// NounPhraseModelV3_81252191306704
// MI455X (gfx1250) — matrix operand order repaired and checked by reading, not compiled, not run on hardware
//
#include <hip/hip_runtime.h>
#include <hip/hip_bf16.h>

// Problem constants (match reference)
#define N_GROUPS 2048
#define LCTX     64
#define NQ       8192
#define DQ       512
#define DK       256
#define DV       256
#define DZ       256

typedef __attribute__((ext_vector_type(16))) __bf16 v16bf;
typedef __attribute__((ext_vector_type(8)))  float  v8f;

__device__ __forceinline__ __bf16 f2bf(float x) { return (__bf16)x; }

#define WMMA_BF16(a, b, c) \
    __builtin_amdgcn_wmma_f32_16x16x32_bf16(false, (a), false, (b), (short)0, (c), false, false)

// ---------------------------------------------------------------------------
// A-fragment (16x32 bf16) from row-major fp32 A with row stride sa.
// ISA 7.12.2: lanes 0-15 -> row M=lane, elem 0..7 = K off+0..7, 8..15 = K off+16..23
//             (off = 8 for lanes 16-31).
// ---------------------------------------------------------------------------
__device__ __forceinline__ v16bf load_a_frag(const float* __restrict__ A, int sa,
                                             int mbase, int kbase, int lane) {
    const int half = lane >> 4;
    const int m    = mbase + (lane & 15);
    const float* row = A + (long)m * sa + kbase + (half << 3);
    v16bf a;
#pragma unroll
    for (int j = 0; j < 4; ++j) {
        float2 p0 = *(const float2*)(row + 2 * j);
        float2 p1 = *(const float2*)(row + 16 + 2 * j);
        a[2 * j]         = f2bf(p0.x);
        a[2 * j + 1]     = f2bf(p0.y);
        a[8 + 2 * j]     = f2bf(p1.x);
        a[8 + 2 * j + 1] = f2bf(p1.y);
    }
    return a;
}

// ---------------------------------------------------------------------------
// Kernel 0: pack Wq (512x256) and Wk (256x256) fp32 -> bf16 B-fragments.
// Fragment layout (matches 16-bit B striping): frag id f = (nt*KS + ks)*32 + lane;
// element e of lane holds B[ks*32 + (lane>=16)*8 + e + (e>=8)*8][nt*16 + (lane&15)].
// One thread per fragment: 16 strided gathers, one 32B contiguous store.
// ---------------------------------------------------------------------------
__global__ __launch_bounds__(256) void pack_b_kernel(const float* __restrict__ Wq,
                                                     const float* __restrict__ Wk,
                                                     __bf16* __restrict__ wqp,
                                                     __bf16* __restrict__ wkp) {
    const int f = blockIdx.x * 256 + threadIdx.x;
    if (f < 16 * 16 * 32) {                 // Wq: 16 n-tiles x 16 k-steps x 32 lanes
        const int lane = f & 31;
        const int ks   = (f >> 5) & 15;
        const int nt   = f >> 9;
        const int n    = nt * 16 + (lane & 15);
        const int K0   = ks * 32 + (lane >> 4) * 8;
        __bf16* dst = wqp + (long)f * 16;
#pragma unroll
        for (int e = 0; e < 16; ++e) dst[e] = f2bf(Wq[(long)(K0 + e + (e >> 3) * 8) * DZ + n]);
    } else if (f < 16 * 16 * 32 + 16 * 8 * 32) {   // Wk: 16 n-tiles x 8 k-steps x 32 lanes
        const int f2   = f - 16 * 16 * 32;
        const int lane = f2 & 31;
        const int ks   = (f2 >> 5) & 7;
        const int nt   = f2 >> 8;
        const int n    = nt * 16 + (lane & 15);
        const int K0   = ks * 32 + (lane >> 4) * 8;
        __bf16* dst = wkp + (long)f2 * 16;
#pragma unroll
        for (int e = 0; e < 16; ++e) dst[e] = f2bf(Wk[(long)(K0 + e + (e >> 3) * 8) * DZ + n]);
    }
}

// ---------------------------------------------------------------------------
// Kernel 1: qz = q @ Wq + bq  (M=8192, K=512, N=256).
// One block per 16-row m-tile. A fragments staged ONCE in LDS as bf16 (16 KB),
// then each of the 8 waves computes 2 n-tiles (32 WMMA) streaming packed B.
// ---------------------------------------------------------------------------
__global__ __launch_bounds__(256) void qz_gemm_kernel(const float* __restrict__ q,
                                                      const __bf16* __restrict__ wqp,
                                                      const float* __restrict__ bq,
                                                      float* __restrict__ qz) {
    __shared__ v16bf afrag[16 * 32];   // [ks][lane] -> 16 KB

    const int tid   = threadIdx.x;
    const int lane  = tid & 31;
    const int wave  = tid >> 5;
    const int mbase = blockIdx.x * 16;

    // ---- stage A fragments (convert fp32 -> bf16 once per m-tile) ----
    for (int f = tid; f < 16 * 32; f += 256) {
        const int fl   = f & 31;
        const int ks   = f >> 5;
        const int mloc = fl & 15;
        const int half = fl >> 4;
        const float* src = q + (long)(mbase + mloc) * DQ + ks * 32 + half * 8;
        float4 a0 = *(const float4*)(src);
        float4 a1 = *(const float4*)(src + 4);
        float4 a2 = *(const float4*)(src + 16);
        float4 a3 = *(const float4*)(src + 20);
        v16bf fr;
        fr[0]  = f2bf(a0.x); fr[1]  = f2bf(a0.y); fr[2]  = f2bf(a0.z); fr[3]  = f2bf(a0.w);
        fr[4]  = f2bf(a1.x); fr[5]  = f2bf(a1.y); fr[6]  = f2bf(a1.z); fr[7]  = f2bf(a1.w);
        fr[8]  = f2bf(a2.x); fr[9]  = f2bf(a2.y); fr[10] = f2bf(a2.z); fr[11] = f2bf(a2.w);
        fr[12] = f2bf(a3.x); fr[13] = f2bf(a3.y); fr[14] = f2bf(a3.z); fr[15] = f2bf(a3.w);
        afrag[f] = fr;
    }
    __syncthreads();

    // ---- each wave: n-tiles wave and wave+8 ----
    const int nt0 = wave, nt1 = wave + 8;
    v8f c0 = {}, c1 = {};
#pragma unroll
    for (int ks = 0; ks < 16; ++ks) {
        v16bf a  = afrag[ks * 32 + lane];
        v16bf b0 = *(const v16bf*)(wqp + (long)((nt0 * 16 + ks) * 32 + lane) * 16);
        v16bf b1 = *(const v16bf*)(wqp + (long)((nt1 * 16 + ks) * 32 + lane) * 16);
        c0 = WMMA_BF16(a, b0, c0);
        c1 = WMMA_BF16(a, b1, c1);
    }

    const int half = lane >> 4;
    const int n0 = nt0 * 16 + (lane & 15);
    const int n1 = nt1 * 16 + (lane & 15);
    const float bias0 = bq[n0], bias1 = bq[n1];
#pragma unroll
    for (int r = 0; r < 8; ++r) {
        const int mm = mbase + r + 8 * half;    // C layout: M = r + 8*(lane>=16)
        qz[(long)mm * DZ + n0] = c0[r] + bias0;
        qz[(long)mm * DZ + n1] = c1[r] + bias1;
    }
}

// ---------------------------------------------------------------------------
// Kernel 2: one block per group g.
//   Phase 1: kz_g = k_g @ Wk + bk (64x256) via WMMA -> LDS (never hits HBM);
//            v_g staged into LDS.
//   Phase 2: per query (binary search on sorted gidx): masked softmax attention.
// LDS rows use stride 257 so (l*257 + d) mod 64 is conflict-free.
// ---------------------------------------------------------------------------
#define KSTR 257

__global__ __launch_bounds__(256) void group_attn_kernel(
        const float* __restrict__ k, const float* __restrict__ v,
        const unsigned char* __restrict__ msk, const int* __restrict__ gidx,
        const __bf16* __restrict__ wkp, const float* __restrict__ bk,
        const float* __restrict__ qz, float* __restrict__ out) {

    __shared__ float kz[LCTX * KSTR];   // 64 KB + pad
    __shared__ float vb[LCTX * KSTR];   // 64 KB + pad
    __shared__ float qbuf[DZ];
    __shared__ float pbuf[256];
    __shared__ float sbuf[LCTX];
    __shared__ float ebuf[LCTX];

    const int g    = blockIdx.x;
    const int tid  = threadIdx.x;
    const int lane = tid & 31;
    const int wave = tid >> 5;

    const float* kg = k + (long)g * LCTX * DK;
    const float* vg = v + (long)g * LCTX * DV;

    // ---- stage v_g into LDS (coalesced global reads) ----
    for (int i = tid; i < LCTX * DV; i += 256) {
        const int l = i >> 8;
        const int d = i & 255;
        vb[l * KSTR + d] = vg[i];
    }

    // ---- phase 1: kz = kg @ Wk + bk via WMMA into LDS ----
    // 64 tiles (4 m x 16 n), 8 waves -> 8 tiles/wave (uniform; EXEC all-ones).
    for (int t = wave; t < 64; t += 8) {
        const int mbase = (t >> 4) * 16;
        const int nt    = t & 15;
        v8f c = {};
#pragma unroll
        for (int ks = 0; ks < 8; ++ks) {
            v16bf a = load_a_frag(kg, DK, mbase, ks * 32, lane);
            v16bf b = *(const v16bf*)(wkp + (long)((nt * 8 + ks) * 32 + lane) * 16);
            c = WMMA_BF16(a, b, c);
        }
        const int half = lane >> 4;
        const int n    = nt * 16 + (lane & 15);
        const float bias = bk[n];
#pragma unroll
        for (int r = 0; r < 8; ++r) {
            const int mm = mbase + r + 8 * half;
            kz[mm * KSTR + n] = c[r] + bias;
        }
    }
    __syncthreads();

    // ---- query range [qs, qe): lower_bound on sorted gidx (redundant/uniform) ----
    int qs, qe;
    {
        int lo = 0, hi = NQ;
        while (lo < hi) { int mid = (lo + hi) >> 1; if (gidx[mid] < g)     lo = mid + 1; else hi = mid; }
        qs = lo;
        hi = NQ;
        while (lo < hi) { int mid = (lo + hi) >> 1; if (gidx[mid] < g + 1) lo = mid + 1; else hi = mid; }
        qe = lo;
    }
    if (qs < qe) __builtin_prefetch(qz + (long)qs * DZ, 0, 1);   // global_prefetch_b8

    const int  lidx  = tid & 63;           // context slot this thread scores
    const int  part  = tid >> 6;           // 4 partial-sum partners per slot
    const bool valid = (msk[g * LCTX + lidx] != 0);
    const float scale = 0.0625f;           // 1/sqrt(DZ)

    for (int qi = qs; qi < qe; ++qi) {
        qbuf[tid] = qz[(long)qi * DZ + tid];
        __syncthreads();

        float partial = 0.f;
        const float* kr = &kz[lidx * KSTR + part * 64];
        const float* qr = &qbuf[part * 64];
#pragma unroll 8
        for (int j = 0; j < 64; ++j) partial += kr[j] * qr[j];
        pbuf[tid] = partial;
        __syncthreads();

        if (tid < 64) {
            const float s = pbuf[tid] + pbuf[tid + 64] + pbuf[tid + 128] + pbuf[tid + 192];
            sbuf[tid] = valid ? s * scale : -3.0e38f;
        }
        __syncthreads();

        float mx = -3.0e38f;
#pragma unroll 8
        for (int l = 0; l < LCTX; ++l) mx = fmaxf(mx, sbuf[l]);

        if (tid < 64) ebuf[tid] = valid ? __expf(sbuf[tid] - mx) : 0.f;
        __syncthreads();

        float sum = 0.f;
#pragma unroll 8
        for (int l = 0; l < LCTX; ++l) sum += ebuf[l];
        const float inv = 1.f / sum;

        float acc = 0.f;
#pragma unroll 8
        for (int l = 0; l < LCTX; ++l) acc += ebuf[l] * vb[l * KSTR + tid];
        out[(long)qi * DV + tid] = acc * inv;
        __syncthreads();
    }
}

// ---------------------------------------------------------------------------
extern "C" void kernel_launch(void* const* d_in, const int* in_sizes, int n_in,
                              void* d_out, int out_size, void* d_ws, size_t ws_size,
                              hipStream_t stream) {
    const float*         q    = (const float*)d_in[0];
    const float*         k    = (const float*)d_in[1];
    const float*         v    = (const float*)d_in[2];
    const unsigned char* m    = (const unsigned char*)d_in[3];  // JAX bool = 1 byte
    const int*           gidx = (const int*)d_in[4];
    const float*         Wq   = (const float*)d_in[5];
    const float*         bq   = (const float*)d_in[6];
    const float*         Wk   = (const float*)d_in[7];
    const float*         bk   = (const float*)d_in[8];
    float* out = (float*)d_out;

    // workspace layout: qz fp32 (8 MB) | Wq packed bf16 (256 KB) | Wk packed bf16 (128 KB)
    float*  qz  = (float*)d_ws;
    __bf16* wqp = (__bf16*)((char*)d_ws + (size_t)NQ * DZ * 4);
    __bf16* wkp = wqp + (size_t)DQ * DZ;

    (void)in_sizes; (void)n_in; (void)out_size; (void)ws_size;

    pack_b_kernel<<<dim3((16 * 16 * 32 + 16 * 8 * 32) / 256), dim3(256), 0, stream>>>(Wq, Wk, wqp, wkp);
    qz_gemm_kernel<<<dim3(NQ / 16), dim3(256), 0, stream>>>(q, wqp, bq, qz);
    group_attn_kernel<<<dim3(N_GROUPS), dim3(256), 0, stream>>>(k, v, m, gidx, wkp, bk, qz, out);
}
